// S2VDQN_layer_12867722019084
// MI455X (gfx1250) — matrix operand order repaired and checked by reading, not compiled, not run on hardware
//
#include <hip/hip_runtime.h>
#include <hip/hip_bf16.h>

typedef __attribute__((ext_vector_type(16))) _Float16 v16h;
typedef __attribute__((ext_vector_type(8)))  float    v8f;

#define N_NODES 50000
#define N_EDGES 800000
#define HID     128

// ---------------- workspace layout (bytes) ----------------
// aggrN : float[50000*128]          @ 0            (25,600,000 B)
// s     : float[50000]              @ 25,600,000   (200,000 B)
// v     : float[128]                @ 25,800,000   (512 B)
// Wt1h  : _Float16[128*32]          @ 25,800,512   (8,192 B)   [n][k], k>=8 zero
// Wt2h  : _Float16[128*128]         @ 25,808,704   (32,768 B)  [n][k]
#define OFF_S    25600000
#define OFF_V    25800000
#define OFF_WT1  25800512
#define OFF_WT2  25808704

// ---- kernel 0: zero aggrN + s (contiguous: 6,450,000 floats) ----
__global__ void zero_ws_kernel(float* __restrict__ p, int n4) {
  float4* p4 = (float4*)p;
  const float4 z = make_float4(0.f, 0.f, 0.f, 0.f);
  for (int i = blockIdx.x * blockDim.x + threadIdx.x; i < n4; i += gridDim.x * blockDim.x)
    p4[i] = z;
}

// ---- kernel 1: weight transpose/convert + v = relu(W4) @ W3 ----
__global__ void prep_kernel(const float* __restrict__ W1, const float* __restrict__ W2,
                            const float* __restrict__ W3, const float* __restrict__ W4,
                            _Float16* __restrict__ Wt1h, _Float16* __restrict__ Wt2h,
                            float* __restrict__ v) {
  const int t = blockIdx.x * blockDim.x + threadIdx.x;
  if (t < HID * HID) {                    // Wt2h[n][k] = W2[k][n]
    const int n = t >> 7, k = t & 127;
    Wt2h[n * HID + k] = (_Float16)W2[k * HID + n];
  }
  if (t < HID * 32) {                     // Wt1h[n][k] = W1[k][n] for k<8, else 0
    const int n = t >> 5, k = t & 31;
    Wt1h[n * 32 + k] = (k < 8) ? (_Float16)W1[k * HID + n] : (_Float16)0.0f;
  }
  if (t < HID) {                          // v[j] = sum_k relu(W4[k]) * W3[k][j]
    float acc = 0.f;
    for (int k = 0; k < HID; ++k) {
      float w = W4[k];
      w = w > 0.f ? w : 0.f;
      acc += w * W3[k * HID + t];
    }
    v[t] = acc;
  }
}

// ---- kernel 2: one wave per edge; coalesced 512B gather + f32 scatter atomics ----
__global__ __launch_bounds__(256) void edge_scatter_kernel(
    const long long* __restrict__ ei, const float* __restrict__ ea,
    const float* __restrict__ nemb, float* __restrict__ aggrN,
    float* __restrict__ s) {
  const int t = blockIdx.x * blockDim.x + threadIdx.x;
  const int e = t >> 5;
  const int lane = t & 31;
  if (e >= N_EDGES) return;
  const int r = (int)ei[e];
  const int c = (int)ei[N_EDGES + e];
  const float4 nb = *(const float4*)(nemb + (size_t)c * HID + lane * 4);
  float* dst = aggrN + (size_t)r * HID + lane * 4;
  atomicAdd(dst + 0, nb.x);
  atomicAdd(dst + 1, nb.y);
  atomicAdd(dst + 2, nb.z);
  atomicAdd(dst + 3, nb.w);
  if (lane == 0) atomicAdd(s + r, ea[e]);
}

// ---- kernel 3: fused WMMA GEMM: out = relu(x@W1 + aggrN@W2 + s.v^T) ----
// Block = 256 threads = 8 waves. Block handles a 16-row M tile; wave w handles
// N tile [16w, 16w+16). Per wave: 5x v_wmma_f32_16x16x32_f16 chained on one v8f acc.
__global__ __launch_bounds__(256) void gemm_kernel(
    const float* __restrict__ x, const float* __restrict__ aggrN,
    const _Float16* __restrict__ Wt1h, const _Float16* __restrict__ Wt2h,
    const float* __restrict__ s, const float* __restrict__ v,
    float* __restrict__ out) {
  const int lane = threadIdx.x & 31;
  const int wv   = threadIdx.x >> 5;      // 0..7  -> N tile
  const int m0   = blockIdx.x << 4;       // 16-row tile (50000 = 3125*16, exact)
  const int mr   = lane & 15;             // A: row; B/C: column within tile
  const int kh   = lane >> 4;             // lane half selector
  const int n    = (wv << 4) + mr;        // global column this lane owns

  v8f acc = {};

  // ---- x @ W1 : K=8 zero-padded to one 32-K WMMA step ----
  v16h Ax = {};
  if (kh == 0) {  // only K=0..7 elements are nonzero (lanes 16..31 hold K>=8 -> 0)
    const float4 x0 = *(const float4*)(x + (size_t)(m0 + mr) * 8);
    const float4 x1 = *(const float4*)(x + (size_t)(m0 + mr) * 8 + 4);
    Ax[0] = (_Float16)x0.x; Ax[1] = (_Float16)x0.y;
    Ax[2] = (_Float16)x0.z; Ax[3] = (_Float16)x0.w;
    Ax[4] = (_Float16)x1.x; Ax[5] = (_Float16)x1.y;
    Ax[6] = (_Float16)x1.z; Ax[7] = (_Float16)x1.w;
  }
  const v16h Bx = *(const v16h*)(Wt1h + n * 32 + kh * 16);  // zeros stored for k>=8
  acc = __builtin_amdgcn_wmma_f32_16x16x32_f16(false, Ax, false, Bx,
                                               (short)0, acc, false, false);

  // ---- aggrN @ W2 : K=128 in four 32-K WMMA steps ----
  const float*    arow = aggrN + (size_t)(m0 + mr) * HID + kh * 16;
  const _Float16* brow = Wt2h + (size_t)n * HID + kh * 16;
#pragma unroll
  for (int kc = 0; kc < HID; kc += 32) {
    const float4 a0 = *(const float4*)(arow + kc);
    const float4 a1 = *(const float4*)(arow + kc + 4);
    const float4 a2 = *(const float4*)(arow + kc + 8);
    const float4 a3 = *(const float4*)(arow + kc + 12);
    v16h A;
    A[0]  = (_Float16)a0.x; A[1]  = (_Float16)a0.y;
    A[2]  = (_Float16)a0.z; A[3]  = (_Float16)a0.w;
    A[4]  = (_Float16)a1.x; A[5]  = (_Float16)a1.y;
    A[6]  = (_Float16)a1.z; A[7]  = (_Float16)a1.w;
    A[8]  = (_Float16)a2.x; A[9]  = (_Float16)a2.y;
    A[10] = (_Float16)a2.z; A[11] = (_Float16)a2.w;
    A[12] = (_Float16)a3.x; A[13] = (_Float16)a3.y;
    A[14] = (_Float16)a3.z; A[15] = (_Float16)a3.w;
    const v16h B = *(const v16h*)(brow + kc);
    acc = __builtin_amdgcn_wmma_f32_16x16x32_f16(false, A, false, B,
                                                 (short)0, acc, false, false);
  }

  // ---- epilogue: + s[m]*v[n], relu, store ----
  const float vn = v[n];
#pragma unroll
  for (int i = 0; i < 8; ++i) {
    const int m = m0 + kh * 8 + i;        // C layout: elem i -> row (kh*8+i), col mr
    float val = acc[i] + s[m] * vn;
    out[(size_t)m * HID + n] = val > 0.f ? val : 0.f;
  }
}

extern "C" void kernel_launch(void* const* d_in, const int* in_sizes, int n_in,
                              void* d_out, int out_size, void* d_ws, size_t ws_size,
                              hipStream_t stream) {
  const float*     x    = (const float*)d_in[0];
  const long long* ei   = (const long long*)d_in[1];   // int64 edge_index [2, E]
  const float*     ea   = (const float*)d_in[2];
  const float*     nemb = (const float*)d_in[3];
  const float*     W1   = (const float*)d_in[4];
  const float*     W2   = (const float*)d_in[5];
  const float*     W3   = (const float*)d_in[6];
  const float*     W4   = (const float*)d_in[7];
  float*           out  = (float*)d_out;

  char* ws = (char*)d_ws;
  float*    aggrN = (float*)ws;
  float*    s     = (float*)(ws + OFF_S);
  float*    v     = (float*)(ws + OFF_V);
  _Float16* Wt1h  = (_Float16*)(ws + OFF_WT1);
  _Float16* Wt2h  = (_Float16*)(ws + OFF_WT2);

  // 1) zero aggrN + s (contiguous 6,450,000 floats = 1,612,500 float4)
  zero_ws_kernel<<<2048, 256, 0, stream>>>(aggrN, 1612500);
  // 2) weight prep (transpose + f16 convert, fold relu(W4)@W3 into v)
  prep_kernel<<<64, 256, 0, stream>>>(W1, W2, W3, W4, Wt1h, Wt2h, v);
  // 3) edge scatter: one wave per edge -> (E*32)/256 blocks
  edge_scatter_kernel<<<(N_EDGES * 32) / 256, 256, 0, stream>>>(ei, ea, nemb, aggrN, s);
  // 4) fused WMMA GEMM + rank-1 + relu epilogue: 50000/16 = 3125 blocks
  gemm_kernel<<<N_NODES / 16, 256, 0, stream>>>(x, aggrN, Wt1h, Wt2h, s, v, out);
}
